// BraidCausalSelfAttention_24507083391053
// MI455X (gfx1250) — matrix operand order repaired and checked by reading, not compiled, not run on hardware
//
#include <hip/hip_runtime.h>

// ---------------------------------------------------------------------------
// CDNA5 (gfx1250) fused causal sigmoid-attention pipeline, bf16 WMMA.
// T=4096, C=1024, H=16, D=64. All matrix math via v_wmma_f32_16x16x32_bf16.
// ---------------------------------------------------------------------------

typedef __attribute__((ext_vector_type(16))) __bf16 bf16x16;
typedef __attribute__((ext_vector_type(8)))  float  f32x8;
typedef __attribute__((ext_vector_type(4)))  float  f32x4;

#define WMMA_BF16(a, b, c) \
  __builtin_amdgcn_wmma_f32_16x16x32_bf16(false, (a), false, (b), (short)0, (c), false, false)

__device__ __forceinline__ float fast_exp2(float x) {
#if __has_builtin(__builtin_amdgcn_exp2f)
  return __builtin_amdgcn_exp2f(x);  // v_exp_f32
#else
  return exp2f(x);
#endif
}
__device__ __forceinline__ float fast_rcp(float x) {
#if __has_builtin(__builtin_amdgcn_rcpf)
  return __builtin_amdgcn_rcpf(x);   // v_rcp_f32
#else
  return 1.0f / x;
#endif
}

// A-fragment (16x32, bf16) from row-major [M,K]-style storage.
// base points at (row m0, col k0); ld = row stride in elements.
// Lane L -> row M = L%16 ; element j: K = (L/16)*16 + j  (16 contiguous).
__device__ __forceinline__ bf16x16 load_a_frag(const __bf16* base, int ld) {
  const int lane = threadIdx.x & 31;
  const __bf16* p = base + (size_t)(lane & 15) * ld + ((lane >> 4) * 16);
  bf16x16 a;
#pragma unroll
  for (int j = 0; j < 16; ++j) a[j] = p[j];
  return a;
}

// B-fragment (32x16, bf16) where the operand is stored as [N, K] row-major
// (K contiguous). base points at (n0, k0); ld = row stride (elements).
// Lane L -> col N = L%16 ; element j: K = (L/16)*16 + j  (16 contiguous).
__device__ __forceinline__ bf16x16 load_b_frag_nk(const __bf16* base, int ld) {
  const int lane = threadIdx.x & 31;
  const __bf16* p = base + (size_t)(lane & 15) * ld + ((lane >> 4) * 16);
  bf16x16 b;
#pragma unroll
  for (int j = 0; j < 16; ++j) b[j] = p[j];
  return b;
}

// ---------------------------------------------------------------------------
// f32 -> bf16 convert (vectorized x4)
// ---------------------------------------------------------------------------
__global__ void cvt_f32_bf16(const float* __restrict__ src,
                             __bf16* __restrict__ dst, int n4) {
  int i = blockIdx.x * blockDim.x + threadIdx.x;
  if (i < n4) {
    f32x4 v = ((const f32x4*)src)[i];
    __bf16* d = dst + (size_t)i * 4;
#pragma unroll
    for (int j = 0; j < 4; ++j) d[j] = (__bf16)v[j];
  }
}

// ---------------------------------------------------------------------------
// GEMM: C[M,N] = A[M,K] * B[N,K]^T  (B stored [N,K] = weight matrix as given)
// Block = 128 threads (4 waves, 2x2), block tile 64x128, wave tile 32x64.
// ---------------------------------------------------------------------------
template <int OUT_BF16>
__global__ __launch_bounds__(128) void gemm_bt(const __bf16* __restrict__ A,
                                               const __bf16* __restrict__ B,
                                               void* __restrict__ Cout,
                                               int M, int N, int K) {
  const int wave = threadIdx.x >> 5;
  const int lane = threadIdx.x & 31;
  const int m0 = blockIdx.x * 64 + (wave & 1) * 32;
  const int n0 = blockIdx.y * 128 + (wave >> 1) * 64;

  f32x8 acc[2][4];
#pragma unroll
  for (int mt = 0; mt < 2; ++mt)
#pragma unroll
    for (int nt = 0; nt < 4; ++nt) acc[mt][nt] = (f32x8)0.0f;

  for (int k0 = 0; k0 < K; k0 += 32) {
    bf16x16 a0 = load_a_frag(A + (size_t)m0 * K + k0, K);
    bf16x16 a1 = load_a_frag(A + (size_t)(m0 + 16) * K + k0, K);
#pragma unroll
    for (int nt = 0; nt < 4; ++nt) {
      bf16x16 b = load_b_frag_nk(B + (size_t)(n0 + nt * 16) * K + k0, K);
      acc[0][nt] = WMMA_BF16(a0, b, acc[0][nt]);
      acc[1][nt] = WMMA_BF16(a1, b, acc[1][nt]);
    }
  }

#pragma unroll
  for (int mt = 0; mt < 2; ++mt)
#pragma unroll
    for (int nt = 0; nt < 4; ++nt)
#pragma unroll
      for (int i = 0; i < 8; ++i) {
        const int r = m0 + mt * 16 + (lane >> 4) * 8 + i;
        const int c = n0 + nt * 16 + (lane & 15);
        const float v = acc[mt][nt][i];
        if (OUT_BF16)
          ((__bf16*)Cout)[(size_t)r * N + c] = (__bf16)v;
        else
          ((float*)Cout)[(size_t)r * N + c] = v;
      }
}

// ---------------------------------------------------------------------------
// RoPE + RMS-norm for q,k ; transpose for v.
// One wave per (t, h). Lane l carries the rotation pair (d=l, d=l+32),
// matching d2 = 32 = wave32 exactly. Cross-lane RMS via __shfl_xor.
//   qtmp/ktmp/vtmp : bf16 [T, C]  (C = H*64)
//   qb, kb         : bf16 [H, T, D]
//   vb             : bf16 [H, D, T]   (transposed for contiguous B-frags)
// ---------------------------------------------------------------------------
__global__ __launch_bounds__(256) void rope_rms_kernel(
    const __bf16* __restrict__ qtmp, const __bf16* __restrict__ ktmp,
    const __bf16* __restrict__ vtmp, const float* __restrict__ cosT,
    const float* __restrict__ sinT, __bf16* __restrict__ qb,
    __bf16* __restrict__ kb, __bf16* __restrict__ vb) {
  const int T = 4096, D = 64;
  const int widx = blockIdx.x * 8 + (threadIdx.x >> 5);
  const int lane = threadIdx.x & 31;
  const int t = widx >> 4;
  const int h = widx & 15;

  const float c = cosT[t * 32 + lane];
  const float s = sinT[t * 32 + lane];
  const size_t ibase = (size_t)t * 1024 + h * 64 + lane;
  const size_t obase = ((size_t)h * T + t) * D + lane;

  // q: rope -> rms
  {
    float x1 = (float)qtmp[ibase], x2 = (float)qtmp[ibase + 32];
    float o1 = x1 * c + x2 * s;
    float o2 = -x1 * s + x2 * c;
    float ss = o1 * o1 + o2 * o2;
#pragma unroll
    for (int m = 16; m >= 1; m >>= 1) ss += __shfl_xor(ss, m, 32);
    float r = rsqrtf(ss * (1.0f / 64.0f) + 1e-6f);
    qb[obase] = (__bf16)(o1 * r);
    qb[obase + 32] = (__bf16)(o2 * r);
  }
  // k: rope -> rms
  {
    float x1 = (float)ktmp[ibase], x2 = (float)ktmp[ibase + 32];
    float o1 = x1 * c + x2 * s;
    float o2 = -x1 * s + x2 * c;
    float ss = o1 * o1 + o2 * o2;
#pragma unroll
    for (int m = 16; m >= 1; m >>= 1) ss += __shfl_xor(ss, m, 32);
    float r = rsqrtf(ss * (1.0f / 64.0f) + 1e-6f);
    kb[obase] = (__bf16)(o1 * r);
    kb[obase + 32] = (__bf16)(o2 * r);
  }
  // v: [T,C] -> [H,D,T] transpose (no rope/rms)
  vb[((size_t)h * D + lane) * T + t] = vtmp[ibase];
  vb[((size_t)h * D + lane + 32) * T + t] = vtmp[ibase + 32];
}

// ---------------------------------------------------------------------------
// Attention chunk body. Processes one 16x64 score chunk for this wave:
// scores (8 WMMAs) -> sigmoid (+causal mask iff MASKED) -> LDS layout swap
// -> y += attn @ v (8 WMMAs). MASKED is only needed on the diagonal chunk;
// bulk chunks are entirely below the diagonal so the predicate is constant.
// ---------------------------------------------------------------------------
template <bool MASKED>
__device__ __forceinline__ void attn_chunk(
    int kbase, const __bf16* __restrict__ kh, const __bf16* __restrict__ vh,
    const bf16x16& qa0, const bf16x16& qa1, __bf16* lds, int lane, int rbase,
    f32x8* yacc) {
  const int T = 4096, D = 64;
  // sigmoid(s/8) = 1 / (1 + exp2(s * -log2(e)/8)); masked -> +huge so
  // exp2 -> inf, rcp(1+inf) -> 0 (== sigmoid(-inf)).
  const float kNegScaleLog2e = -0.125f * 1.44269504f;
  const float kMaskBig = 3.0e38f;

  // --- scores: 16x64 chunk, K=64 -> 4 n-tiles x 2 k-steps = 8 WMMAs ---
  f32x8 sA[4];
#pragma unroll
  for (int nt = 0; nt < 4; ++nt) {
    bf16x16 bk0 = load_b_frag_nk(kh + (size_t)(kbase + nt * 16) * D + 0, D);
    bf16x16 bk1 = load_b_frag_nk(kh + (size_t)(kbase + nt * 16) * D + 32, D);
    f32x8 sacc = (f32x8)0.0f;
    sacc = WMMA_BF16(qa0, bk0, sacc);
    sacc = WMMA_BF16(qa1, bk1, sacc);
    sA[nt] = sacc;
  }
  // --- sigmoid (branchless mask only on diagonal), stash to LDS ---
#pragma unroll
  for (int nt = 0; nt < 4; ++nt)
#pragma unroll
    for (int i = 0; i < 8; ++i) {
      float xe = sA[nt][i] * kNegScaleLog2e;
      if (MASKED) {
        const int col = kbase + nt * 16 + (lane & 15);
        const int r = rbase + i;
        xe = (col <= r) ? xe : kMaskBig;
      }
      const float a = fast_rcp(1.0f + fast_exp2(xe));
      lds[((lane >> 4) * 8 + i) * 64 + nt * 16 + (lane & 15)] = (__bf16)a;
    }
  __syncthreads();  // uniform across waves (chunk counts are block-uniform)
  // --- y += attn(16x64) @ v(64x64) : 2 A-frags, 8 B-frags, 8 WMMAs ---
  const bf16x16 af0 = load_a_frag(lds, 64);
  const bf16x16 af1 = load_a_frag(lds + 32, 64);
#pragma unroll
  for (int nt = 0; nt < 4; ++nt) {
    bf16x16 bv0 = load_b_frag_nk(vh + (size_t)(nt * 16) * T + kbase, T);
    bf16x16 bv1 = load_b_frag_nk(vh + (size_t)(nt * 16) * T + kbase + 32, T);
    yacc[nt] = WMMA_BF16(af0, bv0, yacc[nt]);
    yacc[nt] = WMMA_BF16(af1, bv1, yacc[nt]);
  }
  __syncthreads();
}

// ---------------------------------------------------------------------------
// Attention: per (64 q-rows, head). 4 waves, each owns 16 q rows.
// Bulk chunks (fully below diagonal) run unmasked; the single diagonal
// chunk applies the causal mask. EXEC stays all-ones throughout (WMMA req).
// ---------------------------------------------------------------------------
__global__ __launch_bounds__(128) void attn_kernel(
    const __bf16* __restrict__ qb, const __bf16* __restrict__ kb,
    const __bf16* __restrict__ vb, __bf16* __restrict__ yb) {
  const int T = 4096, D = 64;
  __shared__ __bf16 attn_lds[4][16 * 64];

  const int wave = threadIdx.x >> 5;
  const int lane = threadIdx.x & 31;
  const int h = blockIdx.y;
  const int mblk = blockIdx.x;
  const int m0 = mblk * 64 + wave * 16;

  const __bf16* qh = qb + (size_t)h * T * D;
  const __bf16* kh = kb + (size_t)h * T * D;
  const __bf16* vh = vb + (size_t)h * D * T;

  const bf16x16 qa0 = load_a_frag(qh + (size_t)m0 * D + 0, D);
  const bf16x16 qa1 = load_a_frag(qh + (size_t)m0 * D + 32, D);

  f32x8 yacc[4];
#pragma unroll
  for (int nt = 0; nt < 4; ++nt) yacc[nt] = (f32x8)0.0f;

  const int rbase = m0 + (lane >> 4) * 8;
  __bf16* lds = &attn_lds[wave][0];

  // bulk: chunks fully below the diagonal -> no mask instructions
  for (int kc = 0; kc < mblk; ++kc)
    attn_chunk<false>(kc * 64, kh, vh, qa0, qa1, lds, lane, rbase, yacc);
  // diagonal chunk: causal mask active
  attn_chunk<true>(mblk * 64, kh, vh, qa0, qa1, lds, lane, rbase, yacc);

  const float oscale = 1.0f / (64.0f + 1e-6f);  // 1/(sqrt(T)+1e-6)
#pragma unroll
  for (int nt = 0; nt < 4; ++nt)
#pragma unroll
    for (int i = 0; i < 8; ++i) {
      const int r = m0 + (lane >> 4) * 8 + i;
      const int c = h * 64 + nt * 16 + (lane & 15);
      yb[(size_t)r * 1024 + c] = (__bf16)(yacc[nt][i] * oscale);
    }
}

// ---------------------------------------------------------------------------
// Host-side orchestration
// ---------------------------------------------------------------------------
extern "C" void kernel_launch(void* const* d_in, const int* in_sizes, int n_in,
                              void* d_out, int out_size, void* d_ws,
                              size_t ws_size, hipStream_t stream) {
  const int T = 4096, C = 1024, H = 16;

  const float* x = (const float*)d_in[0];
  const float* cosT = (const float*)d_in[1];
  const float* sinT = (const float*)d_in[2];
  const float* wq = (const float*)d_in[3];
  const float* wk = (const float*)d_in[4];
  const float* wv = (const float*)d_in[5];
  const float* wp = (const float*)d_in[6];

  char* ws = (char*)d_ws;
  auto take = [&](size_t bytes) {
    char* p = ws;
    ws += (bytes + 255) & ~(size_t)255;
    return p;
  };
  __bf16* xb = (__bf16*)take((size_t)T * C * 2);
  __bf16* wqb = (__bf16*)take((size_t)C * C * 2);
  __bf16* wkb = (__bf16*)take((size_t)C * C * 2);
  __bf16* wvb = (__bf16*)take((size_t)C * C * 2);
  __bf16* wpb = (__bf16*)take((size_t)C * C * 2);
  __bf16* qtmp = (__bf16*)take((size_t)T * C * 2);
  __bf16* ktmp = (__bf16*)take((size_t)T * C * 2);
  __bf16* vtmp = (__bf16*)take((size_t)T * C * 2);
  __bf16* qb = (__bf16*)take((size_t)T * C * 2);  // [H,T,64]
  __bf16* kb = (__bf16*)take((size_t)T * C * 2);  // [H,T,64]
  __bf16* vb = (__bf16*)take((size_t)T * C * 2);  // [H,64,T]
  __bf16* yb = (__bf16*)take((size_t)T * C * 2);  // [T,C]

  // 1) f32 -> bf16 converts
  {
    int n4 = (T * C) / 4;
    cvt_f32_bf16<<<(n4 + 255) / 256, 256, 0, stream>>>(x, xb, n4);
    int w4 = (C * C) / 4;
    cvt_f32_bf16<<<(w4 + 255) / 256, 256, 0, stream>>>(wq, wqb, w4);
    cvt_f32_bf16<<<(w4 + 255) / 256, 256, 0, stream>>>(wk, wkb, w4);
    cvt_f32_bf16<<<(w4 + 255) / 256, 256, 0, stream>>>(wv, wvb, w4);
    cvt_f32_bf16<<<(w4 + 255) / 256, 256, 0, stream>>>(wp, wpb, w4);
  }

  // 2) QKV projections: q/k/v = x @ w^T  (bf16 out)
  dim3 gg(T / 64, C / 128);
  gemm_bt<1><<<gg, 128, 0, stream>>>(xb, wqb, qtmp, T, C, C);
  gemm_bt<1><<<gg, 128, 0, stream>>>(xb, wkb, ktmp, T, C, C);
  gemm_bt<1><<<gg, 128, 0, stream>>>(xb, wvb, vtmp, T, C, C);

  // 3) RoPE + RMS norm (q,k) and v transpose
  rope_rms_kernel<<<(T * H) / 8, 256, 0, stream>>>(qtmp, ktmp, vtmp, cosT,
                                                   sinT, qb, kb, vb);

  // 4) causal sigmoid attention
  attn_kernel<<<dim3(T / 64, H), 128, 0, stream>>>(qb, kb, vb, yb);

  // 5) output projection: out = y @ wproj^T (f32 out)
  gemm_bt<0><<<gg, 128, 0, stream>>>(yb, wpb, d_out, T, C, C);
}
